// NAIS_region_distance_Embedding_77395310674423
// MI455X (gfx1250) — matrix operand order repaired and checked by reading, not compiled, not run on hardware
//
#include <hip/hip_runtime.h>
#include <hip/hip_bf16.h>

typedef _Float16 v16h __attribute__((ext_vector_type(16)));
typedef _Float16 v8h  __attribute__((ext_vector_type(8)));
typedef float    v8f  __attribute__((ext_vector_type(8)));
typedef float    v4f  __attribute__((ext_vector_type(4)));

#define HH    200       // history length
#define HPAD  208       // 13 * 16 row tiles
#define NTILE 13        // row tiles per batch
#define DEMB  64        // embed dim (2 x 32)
#define KHID  32        // hidden size

__device__ __forceinline__ float sigmoidf(float x) {
    return 1.0f / (1.0f + __expf(-x));
}

__global__ __launch_bounds__(256)
void nais_region_kernel(const int*   __restrict__ history,
                        const int*   __restrict__ target,
                        const int*   __restrict__ history_region,
                        const int*   __restrict__ target_region,
                        const float* __restrict__ lat_long,
                        const float* __restrict__ W_hist,
                        const float* __restrict__ W_tgt,
                        const float* __restrict__ W_reg,
                        const float* __restrict__ attn1_W,   // [66,32]
                        const float* __restrict__ attn1_b,   // [32]
                        const float* __restrict__ attn2_W,   // [32]
                        const float* __restrict__ dist_W,    // [2,2]
                        const float* __restrict__ dist_b,    // [2]
                        float*       __restrict__ out)
{
    __shared__ __align__(32) _Float16 s_inp[HPAD * DEMB];  // inp = h_emb*t_emb (f16)
    __shared__ __align__(32) _Float16 s_wfrag[2048];       // attn1_W[0:64] B-fragment order
    __shared__ __align__(16) float    s_part[16 * HPAD];   // score partials [n][row]
    __shared__ float    s_temb[DEMB];
    __shared__ float    s_dist[HPAD * 2];
    __shared__ float    s_dot[HPAD];
    __shared__ float    s_w64[KHID], s_w65[KHID], s_bias[KHID], s_a2[KHID];
    __shared__ float    s_red1[256], s_red2[256];

    const int b = blockIdx.x;
    const int t = threadIdx.x;

    // ---------------- Phase 0a: target embedding + weight staging -------------
    if (t < DEMB) {
        float v = (t < 32) ? W_tgt[(size_t)target[b] * 32 + t]
                           : W_reg[(size_t)target_region[b] * 32 + (t - 32)];
        s_temb[t] = v;
    }
    // B fragments: index i = kc*1024 + nt*512 + lane*16 + e
    // element e at lane L holds attn1_W[kc*32 + 8*(L>>4) + (e&7) + 16*(e>>3)][nt*16 + (L&15)]
    for (int i = t; i < 2048; i += 256) {
        int e    = i & 15;
        int ln   = (i >> 4) & 31;
        int nt   = (i >> 9) & 1;
        int kc   = i >> 10;
        int krow = kc * 32 + 8 * (ln >> 4) + (e & 7) + 16 * (e >> 3);
        int col  = 16 * nt + (ln & 15);
        s_wfrag[i] = (_Float16)attn1_W[krow * KHID + col];
    }
    if (t < KHID) {
        s_w64[t]  = attn1_W[64 * KHID + t];
        s_w65[t]  = attn1_W[65 * KHID + t];
        s_bias[t] = attn1_b[t];
        s_a2[t]   = attn2_W[t];
    }
    // dist features: sigmoid((lat_long*100) @ dist_W + dist_b)
    if (t < HH) {
        const float2 ll = ((const float2*)lat_long)[b * HH + t];
        float x0 = ll.x * 100.0f, x1 = ll.y * 100.0f;
        float d0 = x0 * dist_W[0] + x1 * dist_W[2] + dist_b[0];
        float d1 = x0 * dist_W[1] + x1 * dist_W[3] + dist_b[1];
        s_dist[2 * t + 0] = sigmoidf(d0);
        s_dist[2 * t + 1] = sigmoidf(d1);
    } else if (t < HPAD) {
        s_dist[2 * t + 0] = 0.0f;
        s_dist[2 * t + 1] = 0.0f;
        s_dot[t] = 0.0f;
    }
    __syncthreads();

    // ---------------- Phase 0b: gather history embeddings, form inp ----------
    if (t < HH) {
        const float4* hrow = (const float4*)(W_hist + (size_t)history[b * HH + t] * 32);
        const float4* rrow = (const float4*)(W_reg  + (size_t)history_region[b * HH + t] * 32);
        float dotacc = 0.0f;
        _Float16* dst = &s_inp[t * DEMB];
        #pragma unroll
        for (int half = 0; half < 2; ++half) {
            const float4* src = half ? rrow : hrow;
            #pragma unroll
            for (int q = 0; q < 8; ++q) {
                float4 v = src[q];
                int k = half * 32 + 4 * q;
                float p0 = v.x * s_temb[k + 0];
                float p1 = v.y * s_temb[k + 1];
                float p2 = v.z * s_temb[k + 2];
                float p3 = v.w * s_temb[k + 3];
                dotacc += p0 + p1 + p2 + p3;
                dst[k + 0] = (_Float16)p0;
                dst[k + 1] = (_Float16)p1;
                dst[k + 2] = (_Float16)p2;
                dst[k + 3] = (_Float16)p3;
            }
        }
        s_dot[t] = dotacc;
    } else if (t < HPAD) {
        #pragma unroll
        for (int k = 0; k < DEMB; ++k) s_inp[t * DEMB + k] = (_Float16)0.0f;
    }
    __syncthreads();

    // ---------------- Phase 2: WMMA  hid = relu(inp @ W1 + C0), partials of hid@W2
    {
        const int wave = t >> 5;
        const int lane = t & 31;
        const int n    = lane & 15;
        const int hi   = lane >> 4;

        for (int rt = wave; rt < NTILE; rt += 8) {      // wave-uniform loop
            float sacc[8];
            #pragma unroll
            for (int v = 0; v < 8; ++v) sacc[v] = 0.0f;

            #pragma unroll
            for (int nt = 0; nt < 2; ++nt) {
                const int col  = nt * 16 + n;
                const float bc   = s_bias[col];
                const float w64c = s_w64[col];
                const float w65c = s_w65[col];
                const float a2c  = s_a2[col];

                v8f c;
                #pragma unroll
                for (int v = 0; v < 8; ++v) {           // C/D row = v + 8*hi
                    int r = rt * 16 + 8 * hi + v;
                    c[v] = bc + s_dist[2 * r] * w64c + s_dist[2 * r + 1] * w65c;
                }

                #pragma unroll
                for (int kc = 0; kc < 2; ++kc) {
                    // A fragment: row = n; lo halves K=[kc*32+8hi,+8), hi halves +16
                    const _Float16* abase = &s_inp[(rt * 16 + n) * DEMB + kc * 32 + 8 * hi];
                    v8h lo  = *(const v8h*)abase;
                    v8h hi8 = *(const v8h*)(abase + 16);
                    v16h a  = __builtin_shufflevector(lo, hi8,
                              0,1,2,3,4,5,6,7,8,9,10,11,12,13,14,15);
                    v16h bf = *(const v16h*)&s_wfrag[(kc * 2 + nt) * 512 + lane * 16];
                    c = __builtin_amdgcn_wmma_f32_16x16x32_f16(
                            false, a, false, bf, (short)0, c, false, false);
                }

                #pragma unroll
                for (int v = 0; v < 8; ++v) {
                    float hval = fmaxf(c[v], 0.0f);     // relu
                    sacc[v] += hval * a2c;               // hid @ attn2_W partial (2 cols)
                }
            }

            // deposit 8 contiguous row-partials for this column lane:
            // s_part[n][rt*16 + 8*hi + v]  -> two ds_store_b128, no wait chain
            float* pbase = &s_part[n * HPAD + rt * 16 + 8 * hi];
            v4f p0 = { sacc[0], sacc[1], sacc[2], sacc[3] };
            v4f p1 = { sacc[4], sacc[5], sacc[6], sacc[7] };
            *(v4f*)(pbase + 0) = p0;
            *(v4f*)(pbase + 4) = p1;
        }
    }
    __syncthreads();

    // ---------------- Phase 3: row-sum partials, masked exp, beta-norm, sigmoid
    {
        float e = 0.0f, ed = 0.0f;
        if (t < HH) {
            float score = 0.0f;
            #pragma unroll
            for (int n = 0; n < 16; ++n) score += s_part[n * HPAD + t];
            const int tgt = target[b];
            float m  = (history[b * HH + t] != tgt) ? 1.0f : 0.0f;
            float ex = __expf(score) * m;
            e  = ex;
            ed = ex * s_dot[t];
        }
        s_red1[t] = e;
        s_red2[t] = ed;
        __syncthreads();
        #pragma unroll
        for (int s = 128; s > 0; s >>= 1) {
            if (t < s) {
                s_red1[t] += s_red1[t + s];
                s_red2[t] += s_red2[t + s];
            }
            __syncthreads();
        }
        if (t == 0) {
            float pred = s_red2[0] / sqrtf(s_red1[0]);  // beta = 0.5
            out[b] = sigmoidf(pred);
        }
    }
}

extern "C" void kernel_launch(void* const* d_in, const int* in_sizes, int n_in,
                              void* d_out, int out_size, void* d_ws, size_t ws_size,
                              hipStream_t stream) {
    const int*   history        = (const int*)  d_in[0];
    const int*   target         = (const int*)  d_in[1];
    const int*   history_region = (const int*)  d_in[2];
    const int*   target_region  = (const int*)  d_in[3];
    const float* lat_long       = (const float*)d_in[4];
    const float* W_hist         = (const float*)d_in[5];
    const float* W_tgt          = (const float*)d_in[6];
    const float* W_reg          = (const float*)d_in[7];
    const float* attn1_W        = (const float*)d_in[8];
    const float* attn1_b        = (const float*)d_in[9];
    const float* attn2_W        = (const float*)d_in[10];
    const float* dist_W         = (const float*)d_in[11];
    const float* dist_b         = (const float*)d_in[12];
    float* out = (float*)d_out;

    const int B = in_sizes[1];   // number of batches (target count)

    nais_region_kernel<<<dim3(B), dim3(256), 0, stream>>>(
        history, target, history_region, target_region, lat_long,
        W_hist, W_tgt, W_reg, attn1_W, attn1_b, attn2_W, dist_W, dist_b, out);
}
